// Encoding_1039382085701
// MI455X (gfx1250) — matrix operand order repaired and checked by reading, not compiled, not run on hardware
//
#include <hip/hip_runtime.h>
#include <hip/hip_bf16.h>

typedef __attribute__((ext_vector_type(16))) __bf16 v16bf;
typedef __attribute__((ext_vector_type(8)))  __bf16 v8bf;
typedef __attribute__((ext_vector_type(4)))  __bf16 v4bf;
typedef __attribute__((ext_vector_type(8)))  float  v8f;

#define K_CODES 32
#define C_CH    256
#define TILE_N  64
#define TILES_PER_BLK 2
#define BLKS_PER_BATCH 32     // 32 blocks * 2 tiles * 64 tokens = 4096
#define NTOK    4096

// Fused: xc GEMM (WMMA bf16) -> softmax -> aggregation GEMM (WMMA bf16).
// Each block processes TILES_PER_BLK token tiles with GEMM2 accumulators and
// wsum kept in registers across tiles; one global f32 atomic flush at the end.
__global__ __launch_bounds__(256) void enc_fused(const float* __restrict__ x,
                                                 const float* __restrict__ cw,
                                                 const float* __restrict__ scale,
                                                 float* __restrict__ wx_acc,
                                                 float* __restrict__ ws_acc) {
  __shared__ __bf16 xs[C_CH][TILE_N];     // x tile, CHANNEL-major (bf16)
  __shared__ __bf16 cws[K_CODES][C_CH];   // codewords (bf16)
  __shared__ float  xc[TILE_N][K_CODES];  // GEMM1 output staging
  __shared__ __bf16 wT[K_CODES][TILE_N];  // softmax weights, transposed
  __shared__ float  x_sq[TILE_N];
  __shared__ float  c_sq[K_CODES];
  __shared__ float  scl[K_CODES];

  const int t   = threadIdx.x;
  const int b   = blockIdx.x >> 5;        // batch
  const int blk = blockIdx.x & 31;        // block-in-batch
  const float* xb = x + (size_t)b * C_CH * NTOK;

  // ---- stage codewords (bf16), c_sq, scale: once per block ----
  for (int i = t; i < K_CODES * C_CH; i += 256) {
    cws[i >> 8][i & 255] = (__bf16)cw[i];
  }
  if (t < K_CODES) {
    scl[t] = scale[t];
    float s = 0.f;
    #pragma unroll 8
    for (int c = 0; c < C_CH; ++c) { float v = cw[t * C_CH + c]; s += v * v; }
    c_sq[t] = s;
  }

  const int lane = t & 31;
  const int l15  = lane & 15;
  const int hi   = lane >> 4;
  const int wv   = t >> 5;                // wave 0..7

  // persistent GEMM2 state: this wave owns channels [32*wv, 32*wv+32)
  const int ct0 = wv * 2;
  const int ct1 = ct0 + 1;
  const int c0  = ct0 * 16 + l15;
  const int c1  = ct1 * 16 + l15;
  v8f a00 = {}, a01 = {}, a10 = {}, a11 = {};
  float wsum_loc = 0.f;

  for (int tile = 0; tile < TILES_PER_BLK; ++tile) {
    const int n0 = (blk * TILES_PER_BLK + tile) * TILE_N;

    __syncthreads();                      // prior-iter readers done with xs/wT
    if (t < TILE_N) x_sq[t] = 0.0f;
    __syncthreads();

    // ---- stage x tile (coalesced float4 -> packed b64 LDS stores) ----
    {
      const int q = t & 15;               // tokens 4q..4q+3
      const int r = t >> 4;               // channel phase 0..15
      float s0 = 0.f, s1 = 0.f, s2 = 0.f, s3 = 0.f;
      #pragma unroll 4
      for (int it = 0; it < 16; ++it) {
        const int c = r + it * 16;
        const float4 v = *(const float4*)(xb + (size_t)c * NTOK + n0 + 4 * q);
        v4bf pk;
        pk[0] = (__bf16)v.x; pk[1] = (__bf16)v.y;
        pk[2] = (__bf16)v.z; pk[3] = (__bf16)v.w;
        *(v4bf*)&xs[c][4 * q] = pk;       // contiguous 8B store
        s0 += v.x * v.x; s1 += v.y * v.y; s2 += v.z * v.z; s3 += v.w * v.w;
      }
      atomicAdd(&x_sq[4*q+0], s0);
      atomicAdd(&x_sq[4*q+1], s1);
      atomicAdd(&x_sq[4*q+2], s2);
      atomicAdd(&x_sq[4*q+3], s3);
    }
    __syncthreads();

    // ---- GEMM1: xc[n,k] = sum_c xs[c,n] * cws[k,c] ----
    {
      const int mt  = wv >> 1;            // token tile 0..3
      const int nt  = wv & 1;             // code tile 0..1
      const int row = mt * 16 + l15;      // token (A row)
      const int col = nt * 16 + l15;      // code  (B column)
      v8f acc = {};
      #pragma unroll
      for (int ks = 0; ks < 8; ++ks) {
        const int cb = ks * 32;
        v16bf a, bm;
        // A layout: elems 0-7 -> K = cb+8*hi+j ; elems 8-15 -> K = cb+16+8*hi+j
        #pragma unroll
        for (int j = 0; j < 8; ++j) {
          a[j]     = xs[cb + 8 * hi + j][row];
          a[j + 8] = xs[cb + 16 + 8 * hi + j][row];
        }
        // B layout: elems 0-7 -> K = cb+8*hi+j ; elems 8-15 -> K = cb+16+8*hi+j
        v8bf blo = *(const v8bf*)&cws[col][cb + 8 * hi];
        v8bf bhi = *(const v8bf*)&cws[col][cb + 16 + 8 * hi];
        #pragma unroll
        for (int j = 0; j < 8; ++j) { bm[j] = blo[j]; bm[j + 8] = bhi[j]; }
        acc = __builtin_amdgcn_wmma_f32_16x16x32_bf16(false, a, false, bm,
                                                      (short)0, acc, false, false);
      }
      // D layout: VGPR r -> M = r + 8*hi ; column = l15
      #pragma unroll
      for (int r2 = 0; r2 < 8; ++r2)
        xc[mt * 16 + r2 + 8 * hi][nt * 16 + l15] = acc[r2];
    }
    __syncthreads();

    // ---- softmax over K=32 codes per token ----
    if (t < TILE_N) {
      const float xq = x_sq[t];
      float sv[K_CODES];
      float m = -1e30f;
      #pragma unroll
      for (int k = 0; k < K_CODES; ++k) {
        float v = scl[k] * (xq - 2.f * xc[t][k] + c_sq[k]);
        sv[k] = v;
        m = fmaxf(m, v);
      }
      float sum = 0.f;
      #pragma unroll
      for (int k = 0; k < K_CODES; ++k) { float e = __expf(sv[k] - m); sv[k] = e; sum += e; }
      const float inv = 1.f / sum;
      #pragma unroll
      for (int k = 0; k < K_CODES; ++k) wT[k][t] = (__bf16)(sv[k] * inv);
    }
    __syncthreads();

    // ---- wsum[k] partial (register accumulation across tiles) ----
    if (t < K_CODES) {
      float s = 0.f;
      #pragma unroll 8
      for (int n = 0; n < TILE_N; ++n) s += (float)wT[t][n];
      wsum_loc += s;
    }

    // ---- GEMM2: wx[k,c] += sum_n wT[k,n] * xs[c,n] (register accumulators) ----
    #pragma unroll
    for (int s = 0; s < 2; ++s) {
      const int tb = s * 32;              // token step
      v16bf A0, A1, B0, B1;
      {
        // A rows = codes; contiguous token runs in wT
        v8bf lo0 = *(const v8bf*)&wT[l15][tb + 8 * hi];
        v8bf h0  = *(const v8bf*)&wT[l15][tb + 16 + 8 * hi];
        v8bf lo1 = *(const v8bf*)&wT[16 + l15][tb + 8 * hi];
        v8bf h1  = *(const v8bf*)&wT[16 + l15][tb + 16 + 8 * hi];
        #pragma unroll
        for (int j = 0; j < 8; ++j) {
          A0[j] = lo0[j]; A0[j + 8] = h0[j];
          A1[j] = lo1[j]; A1[j + 8] = h1[j];
        }
      }
      {
        // B columns = channels; inner = contiguous tokens in xs[c][*]
        v8bf b0l = *(const v8bf*)&xs[c0][tb + 8 * hi];
        v8bf b0h = *(const v8bf*)&xs[c0][tb + 16 + 8 * hi];
        v8bf b1l = *(const v8bf*)&xs[c1][tb + 8 * hi];
        v8bf b1h = *(const v8bf*)&xs[c1][tb + 16 + 8 * hi];
        #pragma unroll
        for (int j = 0; j < 8; ++j) {
          B0[j] = b0l[j]; B0[j + 8] = b0h[j];
          B1[j] = b1l[j]; B1[j + 8] = b1h[j];
        }
      }
      a00 = __builtin_amdgcn_wmma_f32_16x16x32_bf16(false, A0, false, B0, (short)0, a00, false, false);
      a01 = __builtin_amdgcn_wmma_f32_16x16x32_bf16(false, A0, false, B1, (short)0, a01, false, false);
      a10 = __builtin_amdgcn_wmma_f32_16x16x32_bf16(false, A1, false, B0, (short)0, a10, false, false);
      a11 = __builtin_amdgcn_wmma_f32_16x16x32_bf16(false, A1, false, B1, (short)0, a11, false, false);
    }
  }

  // ---- single flush: global f32 atomic accumulation ----
  {
    float* wxb = wx_acc + (size_t)b * K_CODES * C_CH;
    #pragma unroll
    for (int r2 = 0; r2 < 8; ++r2) {
      const int klo = r2 + 8 * hi;        // codes 0..15 tile
      const int khi = 16 + r2 + 8 * hi;   // codes 16..31 tile
      atomicAdd(&wxb[klo * C_CH + c0], a00[r2]);
      atomicAdd(&wxb[klo * C_CH + c1], a01[r2]);
      atomicAdd(&wxb[khi * C_CH + c0], a10[r2]);
      atomicAdd(&wxb[khi * C_CH + c1], a11[r2]);
    }
    if (t < K_CODES) atomicAdd(&ws_acc[b * K_CODES + t], wsum_loc);
  }
}

// out[b,k,c] = wx[b,k,c] - wsum[b,k] * cw[k,c]
__global__ __launch_bounds__(256) void enc_finalize(const float* __restrict__ wx,
                                                    const float* __restrict__ ws,
                                                    const float* __restrict__ cw,
                                                    float* __restrict__ out) {
  const int idx = blockIdx.x * 256 + threadIdx.x;   // 32*32*256 total
  const int kc  = idx & (K_CODES * C_CH - 1);
  const int bb  = idx >> 13;
  const int k   = kc >> 8;
  out[idx] = wx[idx] - ws[bb * K_CODES + k] * cw[kc];
}

extern "C" void kernel_launch(void* const* d_in, const int* in_sizes, int n_in,
                              void* d_out, int out_size, void* d_ws, size_t ws_size,
                              hipStream_t stream) {
  const float* x     = (const float*)d_in[0];   // (32,256,64,64)
  const float* cw    = (const float*)d_in[1];   // (32,256)
  const float* scale = (const float*)d_in[2];   // (32,)
  float* out = (float*)d_out;                   // (32,32,256)

  float* wx_acc = (float*)d_ws;                        // 32*32*256 f32
  float* ws_acc = wx_acc + 32 * K_CODES * C_CH;        // 32*32 f32

  const size_t zero_bytes = (size_t)(32 * K_CODES * C_CH + 32 * K_CODES) * sizeof(float);
  hipMemsetAsync(d_ws, 0, zero_bytes, stream);

  enc_fused<<<dim3(32 * BLKS_PER_BATCH), dim3(256), 0, stream>>>(x, cw, scale, wx_acc, ws_acc);
  enc_finalize<<<dim3((32 * K_CODES * C_CH) / 256), dim3(256), 0, stream>>>(wx_acc, ws_acc, cw, out);
}
